// ChannelProjector2D_26371099198074
// MI455X (gfx1250) — matrix operand order repaired and checked by reading, not compiled, not run on hardware
//
#include <hip/hip_runtime.h>

// CDNA5 / gfx1250: wave32, WMMA 16x16x32 bf16 (f32 accumulate).
// GEMM: A[401408,256] (x) * B[256,256] (W) -> C[401408,256], all f32 in HBM.
// Precision: split-bf16 (bf16x3): hi = top-16-bit truncation of f32 (exact bf16),
// lo = bf16(residual); C ~= Ah*Bh + Al*Bh + Ah*Bl  (~1e-4 rel, near-fp32).
// Roofline: 822 MB @ 23.3 TB/s ~= 35 us floor; 3x bf16 WMMA work ~= 30 us.

typedef __attribute__((ext_vector_type(16))) __bf16       v16bf;
typedef __attribute__((ext_vector_type(8)))  float        v8f;
typedef __attribute__((ext_vector_type(4)))  float        v4f;
typedef __attribute__((ext_vector_type(4)))  unsigned int v4u;

union FragU { v4u q[2]; v16bf v; };  // 8 dwords <-> 16 bf16 (A fragment)

static __device__ __forceinline__ __bf16 bf16_from_bits(unsigned short b) {
  union { unsigned short s; __bf16 f; } u; u.s = b; return u.f;
}
// Pack two bf16 (RNE) into one dword: a -> [15:0], b -> [31:16].
static __device__ __forceinline__ unsigned pack_bf16_rne(float a, float b) {
  union { struct { __bf16 l, h; } p; unsigned u; } q;
  q.p.l = (__bf16)a; q.p.h = (__bf16)b; return q.u;
}

__global__ __launch_bounds__(512, 1)
void cproj_wmma_kernel(const float* __restrict__ x,
                       const float* __restrict__ W,
                       float* __restrict__ out,
                       int mtiles) {
  constexpr int K = 256;
  constexpr int N = 256;

  const int tid    = threadIdx.x;
  const int lane   = tid & 31;
  const int wave   = tid >> 5;   // 0..15 -> which 16-column N tile
  const int half   = lane >> 4;  // lane half (0/1)
  const int lanelo = lane & 15;
  const int n0     = wave << 4;

  // A-fragment staging: hi & lo planes of one 16x256 tile, in fragment order.
  // uint4 index: kc*64 + (elems8..15 ? 32 : 0) + lane   (lane stride = 16 B).
  __shared__ v4u lds_hi[512];   // 8 KB
  __shared__ v4u lds_lo[512];   // 8 KB

  // ---- Persistent B fragments (hi/lo split) for this wave's 16 columns.
  // Dense 16-bit B 32x16 layout: lane L, elem j <-> K = kc*32 + 2*(j&8) + 8*(L/16) + (j&7),
  // N = n0 + L%16.  W is tiny and L2-hot; one-time build.
  v16bf bhi[8], blo[8];
#pragma unroll
  for (int kc = 0; kc < 8; ++kc) {
#pragma unroll
    for (int j = 0; j < 16; ++j) {
      const int k = kc * 32 + (j & 8) * 2 + half * 8 + (j & 7);
      const float w = W[k * N + n0 + lanelo];
      const unsigned u = __float_as_uint(w);
      bhi[kc][j] = bf16_from_bits((unsigned short)(u >> 16));
      blo[kc][j] = (__bf16)(w - __uint_as_float(u & 0xffff0000u));
    }
  }

  // Writer role: thread stages 8 consecutive k of one row of the A tile.
  const int wr_r    = tid & 15;        // row in tile
  const int wr_kseg = tid >> 4;        // 0..31, k = wr_kseg*8 .. +7
  const int wr_kc   = wr_kseg >> 2;
  const int wr_jh   = (wr_kseg >> 1) & 1;  // frag elems 0-7 vs 8-15
  const int wr_h    = wr_kseg & 1;         // lane half the elems belong to
  const int wr_idx  = wr_kc * 64 + wr_jh * 32 + wr_h * 16 + wr_r;

  const int stride = gridDim.x;
  int mt = blockIdx.x;

  // Register double-buffer of the raw f32 staging data.
  v4f p0 = {0.f, 0.f, 0.f, 0.f}, p1 = {0.f, 0.f, 0.f, 0.f};
  if (mt < mtiles) {
    const float* src = x + (long)(mt * 16 + wr_r) * K + wr_kseg * 8;
    p0 = *(const v4f*)src;
    p1 = *(const v4f*)(src + 4);
  }

  for (; mt < mtiles; mt += stride) {
    // ---- Convert staged 8 floats -> packed hi (trunc) + lo (bf16 of residual).
    v4u hq, lq;
    {
      const unsigned u0 = __float_as_uint(p0[0]), u1 = __float_as_uint(p0[1]);
      const unsigned u2 = __float_as_uint(p0[2]), u3 = __float_as_uint(p0[3]);
      const unsigned u4 = __float_as_uint(p1[0]), u5 = __float_as_uint(p1[1]);
      const unsigned u6 = __float_as_uint(p1[2]), u7 = __float_as_uint(p1[3]);
      hq[0] = (u1 & 0xffff0000u) | (u0 >> 16);   // v_perm-able hi packs
      hq[1] = (u3 & 0xffff0000u) | (u2 >> 16);
      hq[2] = (u5 & 0xffff0000u) | (u4 >> 16);
      hq[3] = (u7 & 0xffff0000u) | (u6 >> 16);
      lq[0] = pack_bf16_rne(p0[0] - __uint_as_float(u0 & 0xffff0000u),
                            p0[1] - __uint_as_float(u1 & 0xffff0000u));
      lq[1] = pack_bf16_rne(p0[2] - __uint_as_float(u2 & 0xffff0000u),
                            p0[3] - __uint_as_float(u3 & 0xffff0000u));
      lq[2] = pack_bf16_rne(p1[0] - __uint_as_float(u4 & 0xffff0000u),
                            p1[1] - __uint_as_float(u5 & 0xffff0000u));
      lq[3] = pack_bf16_rne(p1[2] - __uint_as_float(u6 & 0xffff0000u),
                            p1[3] - __uint_as_float(u7 & 0xffff0000u));
    }

    __syncthreads();                 // previous tile's fragment reads done
    lds_hi[wr_idx] = hq;             // ds_store_b128, stride-16B: conflict-free
    lds_lo[wr_idx] = lq;
    __syncthreads();                 // fragments visible to all waves

    // ---- Prefetch next tile's raw f32 into registers (hidden under compute).
    const int mtn = mt + stride;
    if (mtn < mtiles) {
      const float* src = x + (long)(mtn * 16 + wr_r) * K + wr_kseg * 8;
      p0 = *(const v4f*)src;
      p1 = *(const v4f*)(src + 4);
    }

    // ---- Compute: 16x16 output tile per wave, K=256 in 8 chunks of 32.
    v8f c = {0.f, 0.f, 0.f, 0.f, 0.f, 0.f, 0.f, 0.f};
#pragma unroll
    for (int kc = 0; kc < 8; ++kc) {
      FragU ah, al;
      ah.q[0] = lds_hi[kc * 64 + lane];        // elems 0..7
      ah.q[1] = lds_hi[kc * 64 + 32 + lane];   // elems 8..15
      al.q[0] = lds_lo[kc * 64 + lane];
      al.q[1] = lds_lo[kc * 64 + 32 + lane];
      c = __builtin_amdgcn_wmma_f32_16x16x32_bf16(false, ah.v, false, bhi[kc],
                                                  (short)0, c, false, false);
      c = __builtin_amdgcn_wmma_f32_16x16x32_bf16(false, al.v, false, bhi[kc],
                                                  (short)0, c, false, false);
      c = __builtin_amdgcn_wmma_f32_16x16x32_bf16(false, ah.v, false, blo[kc],
                                                  (short)0, c, false, false);
    }

    // C/D 16x16 f32 layout: VGPR r, lane L -> M = r + 8*(L/16), N = L%16.
    float* obase = out + (long)(mt * 16) * N + n0 + lanelo;
#pragma unroll
    for (int r = 0; r < 8; ++r) {
      __builtin_nontemporal_store(c[r], obase + (long)(r + half * 8) * N);
    }
  }
}

extern "C" void kernel_launch(void* const* d_in, const int* in_sizes, int n_in,
                              void* d_out, int out_size, void* d_ws, size_t ws_size,
                              hipStream_t stream) {
  const float* x = (const float*)d_in[0];   // [8,224,224,256] f32
  const float* W = (const float*)d_in[1];   // [256,256] f32
  float* out = (float*)d_out;               // [8,224,224,256] f32

  const int rows   = in_sizes[0] / 256;     // 401408
  const int mtiles = rows / 16;             // 25088
  int blocks = mtiles < 1568 ? mtiles : 1568;  // 16 M-tiles per block

  cproj_wmma_kernel<<<dim3(blocks), dim3(512), 0, stream>>>(x, W, out, mtiles);
}
